// RPMNet_16123307229295
// MI455X (gfx1250) — matrix operand order repaired and checked by reading, not compiled, not run on hardware
//
#include <hip/hip_runtime.h>
#include <math.h>

// RPMNet early-fusion feature extraction for MI455X (gfx1250, wave32).
// GEMM layers run on v_wmma_f32_16x16x32_f16 with pre-packed f16 operands.

typedef __attribute__((ext_vector_type(16))) _Float16 v16h;
typedef __attribute__((ext_vector_type(8)))  _Float16 v8h;
typedef __attribute__((ext_vector_type(8)))  float    v8f;

#define BATCH   4
#define NPTS    1024
#define NPOINT  512
#define NSAMPLE 64
#define PTOT    (BATCH * NPOINT * NSAMPLE)   // 131072
#define QTOT    (BATCH * NPOINT)             // 2048
#define RADIUS2 0.09f
#define EPS_GN  1e-5f
#define EPS_NRM 1e-5f

// ---------------- Farthest point sampling ----------------
__global__ void fps_kernel(const float* __restrict__ xyz, int* __restrict__ fps)
{
    int b   = blockIdx.x;
    int tid = threadIdx.x;                 // == point index n
    const float* X = xyz + (size_t)b * NPTS * 3;
    float px = X[tid * 3 + 0], py = X[tid * 3 + 1], pz = X[tid * 3 + 2];
    float dmin = 1e10f;

    __shared__ float sv[32];
    __shared__ int   si[32];
    __shared__ int   s_next;

    int cur  = 0;
    int lane = tid & 31;
    int wv   = tid >> 5;

    for (int it = 0; it < NPOINT; ++it) {
        float cx = X[cur * 3 + 0], cy = X[cur * 3 + 1], cz = X[cur * 3 + 2];
        float dx = px - cx, dy = py - cy, dz = pz - cz;
        dmin = fminf(dmin, dx * dx + dy * dy + dz * dz);

        float v = dmin; int idx = tid;
        #pragma unroll
        for (int off = 16; off >= 1; off >>= 1) {
            float ov = __shfl_xor(v, off, 32);
            int   oi = __shfl_xor(idx, off, 32);
            if (ov > v || (ov == v && oi < idx)) { v = ov; idx = oi; }
        }
        if (lane == 0) { sv[wv] = v; si[wv] = idx; }
        __syncthreads();
        if (tid < 32) {
            v = sv[tid]; idx = si[tid];
            #pragma unroll
            for (int off = 16; off >= 1; off >>= 1) {
                float ov = __shfl_xor(v, off, 32);
                int   oi = __shfl_xor(idx, off, 32);
                if (ov > v || (ov == v && oi < idx)) { v = ov; idx = oi; }
            }
            if (tid == 0) s_next = idx;
        }
        if (tid == 0) fps[b * NPOINT + it] = cur;
        __syncthreads();
        cur = s_next;
    }
}

// ---------------- Ball query (+ gather new_xyz) ----------------
__global__ void ball_kernel(const float* __restrict__ xyz, const int* __restrict__ fps,
                            int* __restrict__ ball, float* __restrict__ newxyz)
{
    int q = blockIdx.x * blockDim.x + threadIdx.x;
    if (q >= QTOT) return;
    int b = q / NPOINT;
    const float* X = xyz + (size_t)b * NPTS * 3;
    int self = fps[q];
    float cx = X[self * 3 + 0], cy = X[self * 3 + 1], cz = X[self * 3 + 2];
    newxyz[q * 3 + 0] = cx; newxyz[q * 3 + 1] = cy; newxyz[q * 3 + 2] = cz;

    int* out = ball + (size_t)q * NSAMPLE;
    int cnt = 0;
    for (int n = 0; n < NPTS && cnt < NSAMPLE; ++n) {
        if (n == self) continue;
        float dx = X[n * 3 + 0] - cx, dy = X[n * 3 + 1] - cy, dz = X[n * 3 + 2] - cz;
        if (dx * dx + dy * dy + dz * dz <= RADIUS2) out[cnt++] = n;
    }
    for (; cnt < NSAMPLE; ++cnt) out[cnt] = self;
}

__device__ __forceinline__ float angle3(float ax, float ay, float az,
                                        float bx, float by, float bz)
{
    float cx = ay * bz - az * by;
    float cy = az * bx - ax * bz;
    float cz = ax * by - ay * bx;
    float cn = sqrtf(cx * cx + cy * cy + cz * cz);
    float d  = ax * bx + ay * by + az * bz;
    return atan2f(cn, d);
}

// ---------------- Grouped features: dxyz(3) + ppf(4), layout [c][p] ----------------
__global__ void feat_kernel(const float* __restrict__ xyz, const float* __restrict__ nrm,
                            const int* __restrict__ fps, const int* __restrict__ ball,
                            const float* __restrict__ newxyz, float* __restrict__ feat)
{
    int p = blockIdx.x * blockDim.x + threadIdx.x;
    if (p >= PTOT) return;
    int q = p / NSAMPLE;
    int b = q / NPOINT;
    const float* X  = xyz + (size_t)b * NPTS * 3;
    const float* Nr = nrm + (size_t)b * NPTS * 3;
    int self = fps[q];
    int n    = ball[p];
    float cx = newxyz[q * 3 + 0], cy = newxyz[q * 3 + 1], cz = newxyz[q * 3 + 2];
    float dx = X[n * 3 + 0] - cx, dy = X[n * 3 + 1] - cy, dz = X[n * 3 + 2] - cz;
    float rx = Nr[self * 3 + 0], ry = Nr[self * 3 + 1], rz = Nr[self * 3 + 2];
    float ix = Nr[n * 3 + 0],    iy = Nr[n * 3 + 1],    iz = Nr[n * 3 + 2];

    feat[0 * (size_t)PTOT + p] = dx;
    feat[1 * (size_t)PTOT + p] = dy;
    feat[2 * (size_t)PTOT + p] = dz;
    feat[3 * (size_t)PTOT + p] = angle3(rx, ry, rz, dx, dy, dz);
    feat[4 * (size_t)PTOT + p] = angle3(ix, iy, iz, dx, dy, dz);
    feat[5 * (size_t)PTOT + p] = angle3(rx, ry, rz, ix, iy, iz);
    feat[6 * (size_t)PTOT + p] = sqrtf(dx * dx + dy * dy + dz * dz);
}

// ---------------- conv1 (7 -> 48) ----------------
__global__ void conv1_kernel(const float* __restrict__ feat, const float* __restrict__ w,
                             const float* __restrict__ bias, float* __restrict__ y)
{
    int i = blockIdx.x * blockDim.x + threadIdx.x;     // co * PTOT + p
    if (i >= 48 * PTOT) return;
    int co = i / PTOT, p = i % PTOT;
    float acc = bias[co];
    #pragma unroll
    for (int c = 0; c < 7; ++c)
        acc += w[co * 7 + c] * feat[(size_t)c * PTOT + p];
    y[i] = acc;
}

// ---------------- GroupNorm stats: one block per (b, group) ----------------
__global__ void gn_stats(const float* __restrict__ y, float* __restrict__ mean,
                         float* __restrict__ var, int C, int ntot, int spb)
{
    int bg = blockIdx.x;
    int b = bg >> 3, g = bg & 7;
    int gc = C >> 3;
    long count = (long)gc * spb;
    float s1 = 0.f, s2 = 0.f;
    for (long i = threadIdx.x; i < count; i += blockDim.x) {
        int c  = g * gc + (int)(i / spb);
        int sp = (int)(i % spb);
        float v = y[(size_t)c * ntot + (size_t)b * spb + sp];
        s1 += v; s2 += v * v;
    }
    __shared__ float r1[256], r2[256];
    r1[threadIdx.x] = s1; r2[threadIdx.x] = s2;
    __syncthreads();
    for (int off = 128; off >= 1; off >>= 1) {
        if ((int)threadIdx.x < off) {
            r1[threadIdx.x] += r1[threadIdx.x + off];
            r2[threadIdx.x] += r2[threadIdx.x + off];
        }
        __syncthreads();
    }
    if (threadIdx.x == 0) {
        float m = r1[0] / (float)count;
        mean[bg] = m;
        var[bg]  = r2[0] / (float)count - m * m;
    }
}

// ---------------- GN apply + ReLU, f32 in place (used before maxpool) ----------------
__global__ void gn_apply_f32(float* __restrict__ y, const float* __restrict__ mean,
                             const float* __restrict__ var, const float* __restrict__ gamma,
                             const float* __restrict__ beta, int C, int ntot, int spb)
{
    long i = (long)blockIdx.x * blockDim.x + threadIdx.x;
    if (i >= (long)C * ntot) return;
    int c = (int)(i / ntot), p = (int)(i % ntot);
    int bg = (p / spb) * 8 + c / (C >> 3);
    float v = (y[i] - mean[bg]) * rsqrtf(var[bg] + EPS_GN) * gamma[c] + beta[c];
    y[i] = fmaxf(v, 0.f);
}

// ---------------- GN apply + ReLU -> f16 activation buffer for WMMA GEMM ----------------
__global__ void gn_apply_h(const float* __restrict__ y, const float* __restrict__ mean,
                           const float* __restrict__ var, const float* __restrict__ gamma,
                           const float* __restrict__ beta,
                           _Float16* __restrict__ xh, int C, int ntot, int spb)
{
    long i = (long)blockIdx.x * blockDim.x + threadIdx.x;
    if (i >= (long)C * ntot) return;
    int c = (int)(i / ntot), p = (int)(i % ntot);
    int bg = (p / spb) * 8 + c / (C >> 3);
    float v = (y[i] - mean[bg]) * rsqrtf(var[bg] + EPS_GN) * gamma[c] + beta[c];
    xh[i] = (_Float16)fmaxf(v, 0.f);
}

// ---------------- Weight pack: f32 [M x K] -> f16 [M x KPAD], zero padded ----------------
__global__ void pack_w(const float* __restrict__ w, _Float16* __restrict__ wh,
                       int M, int K, int KPAD)
{
    int i = blockIdx.x * blockDim.x + threadIdx.x;
    if (i >= M * KPAD) return;
    int m = i / KPAD, k = i % KPAD;
    wh[i] = (k < K) ? (_Float16)w[m * K + k] : (_Float16)0.f;
}

__global__ void zero_h(_Float16* __restrict__ p, long n)
{
    long i = (long)blockIdx.x * blockDim.x + threadIdx.x;
    if (i < n) p[i] = (_Float16)0.f;
}

// ---------------- WMMA GEMM: Y[M x Npts] = Wh[M x KPAD] * Xh[KPAD x Npts] + bias --------
// Compile-time shapes; one 16x64 output strip (4 N-tiles) per wave, A reused across tiles.
template<int M, int KPAD>
__global__ void gemm_wmma(const _Float16* __restrict__ Wh, const _Float16* __restrict__ Xh,
                          const float* __restrict__ bias, float* __restrict__ Y, int Npts)
{
    int lane = threadIdx.x & 31;
    int wv   = threadIdx.x >> 5;
    int gw   = blockIdx.x * (blockDim.x >> 5) + wv;
    int tiles_n = Npts >> 6;                    // strips of 64 columns
    int mt = gw / tiles_n, nt = gw % tiles_n;
    if (mt >= (M >> 4)) return;                 // wave-uniform
    int mbase = mt << 4, nbase = nt << 6;
    int half  = lane >> 4;
    int mrow  = mbase + (lane & 15);

    v8f acc0 = {}, acc1 = {}, acc2 = {}, acc3 = {};
    #pragma unroll
    for (int ks = 0; ks < KPAD / 32; ++ks) {
        const int kb = ks << 5;
        // A fragment: lane holds row mrow; two contiguous 8-half chunks per k-step
        const _Float16* wr = Wh + (size_t)mrow * KPAD + kb + 8 * half;
        v8h alo = *(const v8h*)(wr);
        v8h ahi = *(const v8h*)(wr + 16);
        v16h a;
        #pragma unroll
        for (int e = 0; e < 8; ++e) { a[e] = alo[e]; a[8 + e] = ahi[e]; }

        // B fragments: lane = N column; K rows 8*half+e and 16+8*half+e (same order as A)
        const _Float16* xr = Xh + (size_t)(kb + 8 * half) * Npts + nbase + (lane & 15);
        v8f* accs[4] = { &acc0, &acc1, &acc2, &acc3 };
        #pragma unroll
        for (int t = 0; t < 4; ++t) {
            v16h bf;
            #pragma unroll
            for (int e = 0; e < 8; ++e) {
                bf[e]     = xr[(size_t)e * Npts + t * 16];
                bf[8 + e] = xr[(size_t)(16 + e) * Npts + t * 16];
            }
            *accs[t] = __builtin_amdgcn_wmma_f32_16x16x32_f16(false, a, false, bf,
                                                              (short)0, *accs[t], false, false);
        }
    }

    int ncol = nbase + (lane & 15);
    v8f* accs[4] = { &acc0, &acc1, &acc2, &acc3 };
    #pragma unroll
    for (int t = 0; t < 4; ++t)
        #pragma unroll
        for (int r = 0; r < 8; ++r) {
            int m = mbase + r + 8 * half;
            Y[(size_t)m * Npts + ncol + t * 16] = (*accs[t])[r] + bias[m];
        }
}

// ---------------- Max pool over the K=64 neighborhood -> f16 for postpool GEMM ----------
__global__ void maxpool_kernel(const float* __restrict__ x, _Float16* __restrict__ pooled)
{
    int i = blockIdx.x * blockDim.x + threadIdx.x;     // c * QTOT + q
    if (i >= 96 * QTOT) return;
    int c = i / QTOT, q = i % QTOT;
    const float* src = x + (size_t)c * PTOT + (size_t)q * NSAMPLE;
    float m = src[0];
    #pragma unroll 8
    for (int k = 1; k < NSAMPLE; ++k) m = fmaxf(m, src[k]);
    pooled[i] = (_Float16)m;
}

// ---------------- Final per-point L2 normalize -> (B,S,96) ----------------
__global__ void norm_kernel(const float* __restrict__ x, float* __restrict__ out)
{
    int q = blockIdx.x * blockDim.x + threadIdx.x;
    if (q >= QTOT) return;
    float s = 0.f;
    for (int c = 0; c < 96; ++c) { float v = x[(size_t)c * QTOT + q]; s += v * v; }
    float inv = 1.f / (sqrtf(s) + EPS_NRM);
    for (int c = 0; c < 96; ++c)
        out[(size_t)q * 96 + c] = x[(size_t)c * QTOT + q] * inv;
}

extern "C" void kernel_launch(void* const* d_in, const int* in_sizes, int n_in,
                              void* d_out, int out_size, void* d_ws, size_t ws_size,
                              hipStream_t stream)
{
    const float* xyz  = (const float*)d_in[0];
    const float* nrm  = (const float*)d_in[1];
    const float* w1   = (const float*)d_in[2];
    const float* b1   = (const float*)d_in[3];
    const float* g1   = (const float*)d_in[4];
    const float* be1  = (const float*)d_in[5];
    const float* w2   = (const float*)d_in[6];
    const float* b2   = (const float*)d_in[7];
    const float* g2   = (const float*)d_in[8];
    const float* be2  = (const float*)d_in[9];
    const float* w3   = (const float*)d_in[10];
    const float* b3   = (const float*)d_in[11];
    const float* g3   = (const float*)d_in[12];
    const float* be3  = (const float*)d_in[13];
    const float* pw1  = (const float*)d_in[14];
    const float* pb1  = (const float*)d_in[15];
    const float* pg1  = (const float*)d_in[16];
    const float* pbe1 = (const float*)d_in[17];
    const float* pw2  = (const float*)d_in[18];
    const float* pb2  = (const float*)d_in[19];
    const float* pg2  = (const float*)d_in[20];
    const float* pbe2 = (const float*)d_in[21];
    const float* pw3  = (const float*)d_in[22];
    const float* pb3  = (const float*)d_in[23];

    float* out        = (float*)d_out;
    float* newxyz_out = out;                    // (B,S,3)
    float* feat_out   = out + QTOT * 3;         // (B,S,96)

    float* ws = (float*)d_ws;
    size_t off = 0;
    int*      fps     = (int*)(ws + off);      off += QTOT;
    int*      ball    = (int*)(ws + off);      off += PTOT;
    float*    feat7   = ws + off;              off += (size_t)7  * PTOT;
    float*    bufF    = ws + off;              off += (size_t)96 * PTOT;      // f32 y buffer
    _Float16* XhA     = (_Float16*)(ws + off); off += (size_t)32 * PTOT;      // 64 x PTOT f16
    _Float16* XhB     = (_Float16*)(ws + off); off += (size_t)32 * PTOT;      // 64 x PTOT f16
    _Float16* w2h     = (_Float16*)(ws + off); off += (48 * 64) / 2;
    _Float16* w3h     = (_Float16*)(ws + off); off += (96 * 64) / 2;
    _Float16* pw1h    = (_Float16*)(ws + off); off += (96 * 96) / 2;
    _Float16* pw2h    = (_Float16*)(ws + off); off += (96 * 96) / 2;
    _Float16* pw3h    = (_Float16*)(ws + off); off += (96 * 96) / 2;
    _Float16* pooledh = (_Float16*)(ws + off); off += (96 * QTOT) / 2;
    float*    pAf     = ws + off;              off += (size_t)96 * QTOT;
    _Float16* pAh     = (_Float16*)(ws + off); off += (96 * QTOT) / 2;
    float*    pBf     = ws + off;              off += (size_t)96 * QTOT;
    _Float16* pBh     = (_Float16*)(ws + off); off += (96 * QTOT) / 2;
    float*    pCf     = ws + off;              off += (size_t)96 * QTOT;
    float*    mean    = ws + off;              off += 32;
    float*    var     = ws + off;              off += 32;

    // --- prep: pack weights to padded f16, zero activation pad rows ---
    pack_w<<<(48 * 64 + 255) / 256, 256, 0, stream>>>(w2,  w2h,  48, 48, 64);
    pack_w<<<(96 * 64 + 255) / 256, 256, 0, stream>>>(w3,  w3h,  96, 48, 64);
    pack_w<<<(96 * 96 + 255) / 256, 256, 0, stream>>>(pw1, pw1h, 96, 96, 96);
    pack_w<<<(96 * 96 + 255) / 256, 256, 0, stream>>>(pw2, pw2h, 96, 96, 96);
    pack_w<<<(96 * 96 + 255) / 256, 256, 0, stream>>>(pw3, pw3h, 96, 96, 96);
    zero_h<<<(16 * PTOT) / 256, 256, 0, stream>>>(XhA + (size_t)48 * PTOT, (long)16 * PTOT);
    zero_h<<<(16 * PTOT) / 256, 256, 0, stream>>>(XhB + (size_t)48 * PTOT, (long)16 * PTOT);

    // --- geometry ---
    fps_kernel <<<BATCH, 1024, 0, stream>>>(xyz, fps);
    ball_kernel<<<QTOT / 256, 256, 0, stream>>>(xyz, fps, ball, newxyz_out);
    feat_kernel<<<PTOT / 256, 256, 0, stream>>>(xyz, nrm, fps, ball, newxyz_out, feat7);

    const int SPB_PRE = NPOINT * NSAMPLE;   // 32768
    const int SPB_POST = NPOINT;            // 512

    // --- prepool MLP ---
    conv1_kernel<<<48 * PTOT / 256, 256, 0, stream>>>(feat7, w1, b1, bufF);
    gn_stats  <<<32, 256, 0, stream>>>(bufF, mean, var, 48, PTOT, SPB_PRE);
    gn_apply_h<<<48 * PTOT / 256, 256, 0, stream>>>(bufF, mean, var, g1, be1, XhA, 48, PTOT, SPB_PRE);

    gemm_wmma<48, 64><<<(3 * (PTOT / 64)) / 8, 256, 0, stream>>>(w2h, XhA, b2, bufF, PTOT);
    gn_stats  <<<32, 256, 0, stream>>>(bufF, mean, var, 48, PTOT, SPB_PRE);
    gn_apply_h<<<48 * PTOT / 256, 256, 0, stream>>>(bufF, mean, var, g2, be2, XhB, 48, PTOT, SPB_PRE);

    gemm_wmma<96, 64><<<(6 * (PTOT / 64)) / 8, 256, 0, stream>>>(w3h, XhB, b3, bufF, PTOT);
    gn_stats    <<<32, 256, 0, stream>>>(bufF, mean, var, 96, PTOT, SPB_PRE);
    gn_apply_f32<<<96 * PTOT / 256, 256, 0, stream>>>(bufF, mean, var, g3, be3, 96, PTOT, SPB_PRE);

    // --- pool + postpool MLP ---
    maxpool_kernel<<<96 * QTOT / 256, 256, 0, stream>>>(bufF, pooledh);

    gemm_wmma<96, 96><<<(6 * (QTOT / 64)) / 8, 256, 0, stream>>>(pw1h, pooledh, pb1, pAf, QTOT);
    gn_stats  <<<32, 256, 0, stream>>>(pAf, mean, var, 96, QTOT, SPB_POST);
    gn_apply_h<<<96 * QTOT / 256, 256, 0, stream>>>(pAf, mean, var, pg1, pbe1, pAh, 96, QTOT, SPB_POST);

    gemm_wmma<96, 96><<<(6 * (QTOT / 64)) / 8, 256, 0, stream>>>(pw2h, pAh, pb2, pBf, QTOT);
    gn_stats  <<<32, 256, 0, stream>>>(pBf, mean, var, 96, QTOT, SPB_POST);
    gn_apply_h<<<96 * QTOT / 256, 256, 0, stream>>>(pBf, mean, var, pg2, pbe2, pBh, 96, QTOT, SPB_POST);

    gemm_wmma<96, 96><<<(6 * (QTOT / 64)) / 8, 256, 0, stream>>>(pw3h, pBh, pb3, pCf, QTOT);

    norm_kernel<<<QTOT / 256, 256, 0, stream>>>(pCf, feat_out);
}
